// TW_TF_31963146617101
// MI455X (gfx1250) — matrix operand order repaired and checked by reading, not compiled, not run on hardware
//
#include <hip/hip_runtime.h>
#include <hip/hip_bf16.h>
#include <math.h>

#define GNN_IN 256
#define HCDIM  256
#define HIDDIM 512
#define NCLSD  40

typedef __attribute__((ext_vector_type(16))) __bf16 v16bf;
typedef __attribute__((ext_vector_type(4)))  __bf16 v4bf;
typedef __attribute__((ext_vector_type(8)))  float  v8f;

__device__ inline v4bf cvt4(float4 v) {
    v4bf r;
    r[0] = (__bf16)v.x; r[1] = (__bf16)v.y;
    r[2] = (__bf16)v.z; r[3] = (__bf16)v.w;
    return r;
}

// ---------------------------------------------------------------------------
// Tiled WMMA GEMM: C[M x Nout] = A[M x K] * W[Nout x K]^T + bias, optional relu
// Block = 256 threads = 8 waves; block tile = 128(M) x 64(N); K step = 32.
// Each wave: one 16x16 M-tile x four N-tiles -> 4 WMMAs / K-step, A reused.
// Software pipeline with LDS ping-pong (single barrier per K-step).
// sched_group_barrier pins the schedule: all 10 ds_loads batch before the
// 4 WMMAs so the waits are partial and loads overlap the matrix pipe.
// ---------------------------------------------------------------------------
__global__ __launch_bounds__(256)
void gemm_wmma_bf16(const float* __restrict__ A, const float* __restrict__ W,
                    const float* __restrict__ bias, float* __restrict__ C,
                    int M, int K, int Nout, int relu)
{
    __shared__ __bf16 As[2][128 * 32];   // 2 x 8 KB
    __shared__ __bf16 Ws[2][64 * 32];    // 2 x 4 KB

    const int tid  = threadIdx.x;
    const int wave = tid >> 5;
    const int lane = tid & 31;
    const int tileM0 = blockIdx.x * 128;
    const int tileN0 = blockIdx.y * 64;

    // A staging: 128x32 floats = 1024 float4, 4 per thread.
    // W staging:  64x32 floats =  512 float4, 2 per thread.
    const int srow  = tid >> 3;        // 0..31, advances by 32 per r
    const int scol4 = tid & 7;

    const bool fast = (tileM0 + 128 <= M) && (tileN0 + 64 <= Nout);

    const int half = lane >> 4;
    const int mloc = (wave << 4) + (lane & 15);
    const int nloc = lane & 15;

    float4 ra[4], rw[2];

    auto load_fast = [&](int k0) {
        #pragma unroll
        for (int r = 0; r < 4; ++r)
            ra[r] = *(const float4*)(A + (size_t)(tileM0 + srow + 32 * r) * K + k0 + scol4 * 4);
        #pragma unroll
        for (int r = 0; r < 2; ++r)
            rw[r] = *(const float4*)(W + (size_t)(tileN0 + srow + 32 * r) * K + k0 + scol4 * 4);
    };
    auto load_slow = [&](int k0) {
        #pragma unroll
        for (int r = 0; r < 4; ++r) {
            int gm = tileM0 + srow + 32 * r;
            ra[r] = make_float4(0.f, 0.f, 0.f, 0.f);
            if (gm < M)
                ra[r] = *(const float4*)(A + (size_t)gm * K + k0 + scol4 * 4);
        }
        #pragma unroll
        for (int r = 0; r < 2; ++r) {
            int gn = tileN0 + srow + 32 * r;
            rw[r] = make_float4(0.f, 0.f, 0.f, 0.f);
            if (gn < Nout)
                rw[r] = *(const float4*)(W + (size_t)gn * K + k0 + scol4 * 4);
        }
    };
    auto store_lds = [&](int buf) {
        #pragma unroll
        for (int r = 0; r < 4; ++r)
            *(v4bf*)&As[buf][(srow + 32 * r) * 32 + scol4 * 4] = cvt4(ra[r]);
        #pragma unroll
        for (int r = 0; r < 2; ++r)
            *(v4bf*)&Ws[buf][(srow + 32 * r) * 32 + scol4 * 4] = cvt4(rw[r]);
    };

    v8f acc0 = {}, acc1 = {}, acc2 = {}, acc3 = {};

    auto frag_a = [&](int buf) {
        v16bf a;
        #pragma unroll
        for (int j = 0; j < 8; ++j) {
            int grp = j >> 2, jj = j & 3;
            int ka  = grp * 16 + half * 8 + jj * 2;
            a[2 * j]     = As[buf][mloc * 32 + ka];
            a[2 * j + 1] = As[buf][mloc * 32 + ka + 1];
        }
        return a;
    };
    auto frag_b = [&](int buf, int nt) {
        v16bf b;
        #pragma unroll
        for (int j = 0; j < 8; ++j) {
            int kb = (j >> 2) * 16 + half * 8 + (j & 3) * 2;
            b[2 * j]     = Ws[buf][(nt * 16 + nloc) * 32 + kb];
            b[2 * j + 1] = Ws[buf][(nt * 16 + nloc) * 32 + kb + 1];
        }
        return b;
    };

    auto compute = [&](int buf) {
        v16bf a  = frag_a(buf);
        v16bf b0 = frag_b(buf, 0);
        v16bf b1 = frag_b(buf, 1);
        v16bf b2 = frag_b(buf, 2);
        v16bf b3 = frag_b(buf, 3);
        acc0 = __builtin_amdgcn_wmma_f32_16x16x32_bf16(false, a, false, b0, (short)0, acc0, false, false);
        acc1 = __builtin_amdgcn_wmma_f32_16x16x32_bf16(false, a, false, b1, (short)0, acc1, false, false);
        acc2 = __builtin_amdgcn_wmma_f32_16x16x32_bf16(false, a, false, b2, (short)0, acc2, false, false);
        acc3 = __builtin_amdgcn_wmma_f32_16x16x32_bf16(false, a, false, b3, (short)0, acc3, false, false);
        // Pin the schedule: batch all DS reads (mask 0x100) ahead of the
        // 4 matrix ops (mask 0x8) so dscnt waits are partial.
        __builtin_amdgcn_sched_group_barrier(0x100, 10, 0);
        __builtin_amdgcn_sched_group_barrier(0x008, 4, 0);
    };

    if (fast) {
        load_fast(0);
        store_lds(0);
        __syncthreads();
        int buf = 0;
        for (int k0 = 0; k0 < K; k0 += 32) {
            bool more = (k0 + 32 < K);
            if (more) load_fast(k0 + 32);   // issue next tile's global loads
            compute(buf);                   // WMMAs hide the load latency
            if (more) store_lds(buf ^ 1);   // loadcnt wait lands here
            __syncthreads();
            buf ^= 1;
        }
    } else {
        load_slow(0);
        store_lds(0);
        __syncthreads();
        int buf = 0;
        for (int k0 = 0; k0 < K; k0 += 32) {
            bool more = (k0 + 32 < K);
            if (more) load_slow(k0 + 32);
            compute(buf);
            if (more) store_lds(buf ^ 1);
            __syncthreads();
            buf ^= 1;
        }
    }

    // Epilogue: D layout -> VGPR i holds M = i + 8*(lane/16), N = lane%16
    #pragma unroll
    for (int nt = 0; nt < 4; ++nt) {
        int n = tileN0 + nt * 16 + nloc;
        if (n < Nout) {
            float bn = bias ? bias[n] : 0.0f;
            const v8f& acc = (nt == 0) ? acc0 : (nt == 1) ? acc1 : (nt == 2) ? acc2 : acc3;
            #pragma unroll
            for (int i = 0; i < 8; ++i) {
                int m = tileM0 + (wave << 4) + i + 8 * half;
                if (m < M) {
                    float v = acc[i] + bn;
                    if (relu) v = fmaxf(v, 0.0f);
                    C[(size_t)m * Nout + n] = v;
                }
            }
        }
    }
}

// ---------------------------------------------------------------------------
// Attention edge kernels
// ---------------------------------------------------------------------------
__global__ __launch_bounds__(256)
void edge_logits_k(const float* __restrict__ q, const float* __restrict__ kk,
                   const int* __restrict__ ei, float* __restrict__ logits, int E_)
{
    int wid  = (int)((blockIdx.x * 256u + threadIdx.x) >> 5);
    int lane = threadIdx.x & 31;
    if (wid >= E_) return;
    int src = ei[wid];
    int dst = ei[E_ + wid];
    const float* qd = q  + (size_t)dst * HCDIM;
    const float* ks = kk + (size_t)src * HCDIM;
    float s0 = 0.f, s1 = 0.f;
    #pragma unroll
    for (int i = 0; i < 4; ++i) {
        int d = lane + 32 * i;
        s0 += qd[d]       * ks[d];
        s1 += qd[128 + d] * ks[128 + d];
    }
    #pragma unroll
    for (int off = 16; off > 0; off >>= 1) {
        s0 += __shfl_down(s0, off, 32);
        s1 += __shfl_down(s1, off, 32);
    }
    if (lane == 0) {
        const float sc = 0.08838834764831845f;  // 1/sqrt(128)
        logits[(size_t)wid * 2]     = s0 * sc;
        logits[(size_t)wid * 2 + 1] = s1 * sc;
    }
}

__device__ inline void atomicMaxF(float* addr, float val) {
    if (val >= 0.0f) atomicMax((int*)addr, __float_as_int(val));
    else             atomicMin((unsigned int*)addr, (unsigned int)__float_as_int(val));
}

__global__ void init_seg_k(float* amax, float* denom, int n2) {
    int i = blockIdx.x * blockDim.x + threadIdx.x;
    if (i < n2) { amax[i] = -INFINITY; denom[i] = 0.0f; }
}

__global__ void seg_max_k(const float* __restrict__ logits, const int* __restrict__ ei,
                          float* __restrict__ amax, int E_) {
    int e = blockIdx.x * blockDim.x + threadIdx.x;
    if (e >= E_) return;
    int dst = ei[E_ + e];
    atomicMaxF(&amax[dst * 2],     logits[(size_t)e * 2]);
    atomicMaxF(&amax[dst * 2 + 1], logits[(size_t)e * 2 + 1]);
}

__global__ void amax_fix_k(float* amax, int n2) {
    int i = blockIdx.x * blockDim.x + threadIdx.x;
    if (i < n2 && !isfinite(amax[i])) amax[i] = 0.0f;
}

__global__ void edge_exp_k(float* __restrict__ logits, const int* __restrict__ ei,
                           const float* __restrict__ amax, float* __restrict__ denom, int E_) {
    int e = blockIdx.x * blockDim.x + threadIdx.x;
    if (e >= E_) return;
    int dst = ei[E_ + e];
    float e0 = expf(logits[(size_t)e * 2]     - amax[dst * 2]);
    float e1 = expf(logits[(size_t)e * 2 + 1] - amax[dst * 2 + 1]);
    logits[(size_t)e * 2]     = e0;
    logits[(size_t)e * 2 + 1] = e1;
    atomicAdd(&denom[dst * 2],     e0);
    atomicAdd(&denom[dst * 2 + 1], e1);
}

__global__ __launch_bounds__(256)
void scatter_msg_k(const float* __restrict__ v, const float* __restrict__ ex,
                   const int* __restrict__ ei, const float* __restrict__ denom,
                   float* __restrict__ hg, int E_)
{
    int wid  = (int)((blockIdx.x * 256u + threadIdx.x) >> 5);
    int lane = threadIdx.x & 31;
    if (wid >= E_) return;
    int src = ei[wid];
    int dst = ei[E_ + wid];
    float a0 = ex[(size_t)wid * 2]     / fmaxf(denom[dst * 2],     1e-16f);
    float a1 = ex[(size_t)wid * 2 + 1] / fmaxf(denom[dst * 2 + 1], 1e-16f);
    const float* vs = v  + (size_t)src * HCDIM;
    float*       hd = hg + (size_t)dst * HCDIM;
    #pragma unroll
    for (int i = 0; i < 4; ++i) {
        int d = lane + 32 * i;
        atomicAdd(&hd[d],       vs[d]       * a0);
        atomicAdd(&hd[128 + d], vs[128 + d] * a1);
    }
}

// ---------------------------------------------------------------------------
// BatchNorm stats + BN/sigmoid, fusion
// ---------------------------------------------------------------------------
__global__ __launch_bounds__(256)
void col_stats_k(const float* __restrict__ X, float* __restrict__ mean,
                 float* __restrict__ var, int M, int C) {
    __shared__ float ssum[256];
    __shared__ float ssq[256];
    int c = blockIdx.x;
    int t = threadIdx.x;
    float s = 0.f, qq = 0.f;
    for (int r = t; r < M; r += 256) {
        float x = X[(size_t)r * C + c];
        s += x; qq += x * x;
    }
    ssum[t] = s; ssq[t] = qq;
    __syncthreads();
    for (int off = 128; off > 0; off >>= 1) {
        if (t < off) { ssum[t] += ssum[t + off]; ssq[t] += ssq[t + off]; }
        __syncthreads();
    }
    if (t == 0) {
        float mu = ssum[0] / (float)M;
        mean[c] = mu;
        var[c]  = ssq[0] / (float)M - mu * mu;
    }
}

__global__ void bn_sigmoid_k(float* __restrict__ X, const float* __restrict__ mean,
                             const float* __restrict__ var, const float* __restrict__ g,
                             const float* __restrict__ be, size_t total, int C) {
    size_t i = (size_t)blockIdx.x * blockDim.x + threadIdx.x;
    if (i >= total) return;
    int c = (int)(i % C);
    float x = (X[i] - mean[c]) * rsqrtf(var[c] + 1e-5f) * g[c] + be[c];
    X[i] = 1.0f / (1.0f + expf(-x));
}

__global__ void fuse_k(float* __restrict__ hg, const float* __restrict__ hm,
                       const float* __restrict__ alpha, const float* __restrict__ beta,
                       size_t total) {
    size_t i = (size_t)blockIdx.x * blockDim.x + threadIdx.x;
    if (i >= total) return;
    hg[i] = alpha[0] * hg[i] + beta[0] * hm[i];
}

// ---------------------------------------------------------------------------
// Launch
// ---------------------------------------------------------------------------
extern "C" void kernel_launch(void* const* d_in, const int* in_sizes, int n_in,
                              void* d_out, int out_size, void* d_ws, size_t ws_size,
                              hipStream_t stream) {
    const float* x_g   = (const float*)d_in[0];
    const int*   ei    = (const int*)  d_in[1];
    const float* x_m   = (const float*)d_in[2];
    const float* Wq    = (const float*)d_in[3];  const float* bq  = (const float*)d_in[4];
    const float* Wk    = (const float*)d_in[5];  const float* bk  = (const float*)d_in[6];
    const float* Wv    = (const float*)d_in[7];  const float* bv  = (const float*)d_in[8];
    const float* Wskip = (const float*)d_in[9];  const float* bsk = (const float*)d_in[10];
    const float* W1    = (const float*)d_in[11]; const float* b1  = (const float*)d_in[12];
    const float* g1    = (const float*)d_in[13]; const float* be1 = (const float*)d_in[14];
    const float* W2    = (const float*)d_in[15]; const float* b2  = (const float*)d_in[16];
    const float* g2    = (const float*)d_in[17]; const float* be2 = (const float*)d_in[18];
    const float* W3    = (const float*)d_in[19]; const float* b3  = (const float*)d_in[20];
    const float* alpha = (const float*)d_in[21]; const float* beta= (const float*)d_in[22];
    const float* Wc1   = (const float*)d_in[23]; const float* bc1 = (const float*)d_in[24];
    const float* Wc2   = (const float*)d_in[25]; const float* bc2 = (const float*)d_in[26];

    const int N_ = in_sizes[0] / GNN_IN;
    const int E_ = in_sizes[1] / 2;
    const size_t n256 = (size_t)N_ * 256;
    const size_t n512 = (size_t)N_ * 512;

    float* ws  = (float*)d_ws;
    // Buffer plan (with reuse):
    //  buf0 [n512]: q (lo half) + k (hi half)  ->  h1  ->  hc1
    //  buf1 [n512]: v (lo half)                ->  h2
    //  hg   [n256], hm [n256], logits/ex [E*2], amax [N*2], denom [N*2], stats
    float* qb    = ws;
    float* kb    = qb + n256;
    float* vb    = ws + n512;
    float* hg    = ws + 2 * n512;
    float* hm    = hg + n256;
    float* lg    = hm + n256;
    float* amax  = lg + (size_t)E_ * 2;
    float* denom = amax + (size_t)N_ * 2;
    float* mean  = denom + (size_t)N_ * 2;
    float* var   = mean + 512;
    float* h1    = ws;              // alias buf0
    float* h2    = ws + n512;       // alias buf1
    float* hc1   = ws;              // alias buf0

    auto gemm = [&](const float* A, const float* W, const float* b, float* C,
                    int M, int K, int Nn, int relu) {
        dim3 grid((M + 127) / 128, (Nn + 63) / 64);
        gemm_wmma_bf16<<<grid, dim3(256), 0, stream>>>(A, W, b, C, M, K, Nn, relu);
    };

    // ---- GNN branch: Q/K/V/skip projections ----
    gemm(x_g, Wq,    bq,  qb, N_, GNN_IN, HCDIM, 0);
    gemm(x_g, Wk,    bk,  kb, N_, GNN_IN, HCDIM, 0);
    gemm(x_g, Wv,    bv,  vb, N_, GNN_IN, HCDIM, 0);
    gemm(x_g, Wskip, bsk, hg, N_, GNN_IN, HCDIM, 0);   // hg starts as skip

    // ---- Attention (segment softmax + scatter) ----
    int n2 = N_ * 2;
    init_seg_k<<<(n2 + 255) / 256, 256, 0, stream>>>(amax, denom, n2);
    int ewaves = (E_ + 7) / 8;                         // 8 waves per 256-block
    edge_logits_k<<<ewaves, 256, 0, stream>>>(qb, kb, ei, lg, E_);
    seg_max_k<<<(E_ + 255) / 256, 256, 0, stream>>>(lg, ei, amax, E_);
    amax_fix_k<<<(n2 + 255) / 256, 256, 0, stream>>>(amax, n2);
    edge_exp_k<<<(E_ + 255) / 256, 256, 0, stream>>>(lg, ei, amax, denom, E_);
    scatter_msg_k<<<ewaves, 256, 0, stream>>>(vb, lg, ei, denom, hg, E_);
    // hg now = h_g ; q/k/v buffers free

    // ---- MLP branch ----
    gemm(x_m, W1, b1, h1, N_, GNN_IN, HIDDIM, 0);
    col_stats_k<<<HIDDIM, 256, 0, stream>>>(h1, mean, var, N_, HIDDIM);
    bn_sigmoid_k<<<(unsigned)((n512 + 255) / 256), 256, 0, stream>>>(h1, mean, var, g1, be1, n512, HIDDIM);

    gemm(h1, W2, b2, h2, N_, HIDDIM, HIDDIM, 0);
    col_stats_k<<<HIDDIM, 256, 0, stream>>>(h2, mean, var, N_, HIDDIM);
    bn_sigmoid_k<<<(unsigned)((n512 + 255) / 256), 256, 0, stream>>>(h2, mean, var, g2, be2, n512, HIDDIM);

    gemm(h2, W3, b3, hm, N_, HIDDIM, HCDIM, 0);

    // ---- Fuse + classifier head ----
    fuse_k<<<(unsigned)((n256 + 255) / 256), 256, 0, stream>>>(hg, hm, alpha, beta, n256);
    gemm(hg, Wc1, bc1, hc1, N_, HCDIM, HIDDIM, 1);     // relu
    gemm(hc1, Wc2, bc2, (float*)d_out, N_, HIDDIM, NCLSD, 0);
}
